// DiT_Llama_833223655518
// MI455X (gfx1250) — matrix operand order repaired and checked by reading, not compiled, not run on hardware
//
#include <hip/hip_runtime.h>
#include <hip/hip_bf16.h>
#include <stdint.h>

// ---------------- model constants ----------------
#define DIMM    1024
#define HEADS   16
#define HD      64
#define LAYERS  6
#define TOKN    1024          // tokens per image (32x32 patches)
#define BATCH   4
#define FFD     2816
#define CONV_C  512
#define IMG     64
#define CINCH   4
#define GN_G    32

typedef __bf16 bf16_t;
typedef bf16_t bf16x16 __attribute__((ext_vector_type(16)));
typedef float  floatx8 __attribute__((ext_vector_type(8)));

// -------- helpers: f32 -> bf16 (round-to-nearest-even), packing --------
static __device__ inline uint16_t f2bf(float f) {
  union { float f; uint32_t u; } c{f};
  return (uint16_t)((c.u + 0x7FFFu + ((c.u >> 16) & 1u)) >> 16);
}
static __device__ inline uint32_t pack_bf16x2(float lo, float hi) {
  return (uint32_t)f2bf(lo) | ((uint32_t)f2bf(hi) << 16);
}
static __device__ inline float silu_f(float x) { return x / (1.0f + __expf(-x)); }

// async copy of 16 bytes/lane from global to LDS (gfx1250 ASYNCcnt path)
static __device__ inline void async_g2l_b128(uint32_t lds_off, const void* gaddr) {
  asm volatile("global_load_async_to_lds_b128 %0, %1, off"
               :: "v"(lds_off), "v"(gaddr) : "memory");
}
static __device__ inline void wait_async_le1() {
  asm volatile("s_wait_asynccnt 0x1" ::: "memory");
}
static __device__ inline void wait_async_0() {
  asm volatile("s_wait_asynccnt 0x0" ::: "memory");
}

#define BM 128
#define BN 128
#define BK 32
#define LDT (BK + 4)   // +4 bf16 pad -> conflict-free fragment reads

// 8x wmma on the current LDS tiles (wave = 64x32 = 4x2 16x16 tiles).
__attribute__((always_inline))
static __device__ inline void wmma_tiles(const uint16_t (&As)[BM][LDT],
                                         const uint16_t (&Bs)[BN][LDT],
                                         floatx8 (&acc)[4][2],
                                         int wr, int wc, int lane)
{
  bf16x16 afrag[4], bfrag[2];
  const int mloc = lane & 15;
  const int kb   = (lane >> 4) * 16;      // A: lane-half selects K 0..15 / 16..31
  #pragma unroll
  for (int ri = 0; ri < 4; ++ri) {
    int r = wr * 64 + ri * 16 + mloc;
    union { uint32_t u[8]; bf16x16 v; } fa;
    #pragma unroll
    for (int j = 0; j < 4; ++j) {
      fa.u[j]     = *(const uint32_t*)&As[r][kb + 2 * j];
      fa.u[4 + j] = *(const uint32_t*)&As[r][8 + kb + 2 * j];
    }
    afrag[ri] = fa.v;
  }
  const int kbb = (lane >> 4) * 16;       // B: lane-half selects K 0..15 / 16..31
  #pragma unroll
  for (int cj = 0; cj < 2; ++cj) {
    int r = wc * 32 + cj * 16 + mloc;
    union { uint32_t u[8]; bf16x16 v; } fb;
    #pragma unroll
    for (int j = 0; j < 8; ++j)
      fb.u[j] = *(const uint32_t*)&Bs[r][kbb + 2 * j];
    bfrag[cj] = fb.v;
  }
  #pragma unroll
  for (int ri = 0; ri < 4; ++ri)
    #pragma unroll
    for (int cj = 0; cj < 2; ++cj)
      acc[ri][cj] = __builtin_amdgcn_wmma_f32_16x16x32_bf16(
          false, afrag[ri], false, bfrag[cj], (short)0, acc[ri][cj], false, false);
}

// =====================================================================
// Generic bf16-WMMA GEMM:  C[M,N] = epilogue( A[M,K] @ W[N,K]^T )
// f32 operands converted to bf16 in-register while staging to LDS.
// Register double-buffering + global_prefetch_b8 two tiles ahead.
// =====================================================================
__global__ __launch_bounds__(256) void gemm_bf16_kernel(
    const float* __restrict__ A, const float* __restrict__ W,
    float* __restrict__ C, int M, int N, int K,
    const float* __restrict__ bias, int act,
    const float* __restrict__ gate, int gateStride, int Tb,
    const float* __restrict__ resid)
{
  __shared__ uint16_t As[BM][LDT];
  __shared__ uint16_t Bs[BN][LDT];
  const int tid  = threadIdx.x;
  const int lane = tid & 31;
  const int wave = tid >> 5;
  const int wr = wave >> 2;            // 0..1
  const int wc = wave & 3;             // 0..3
  const int m0 = blockIdx.y * BM;
  const int n0 = blockIdx.x * BN;

  floatx8 acc[4][2];
  #pragma unroll
  for (int i = 0; i < 4; ++i)
    #pragma unroll
    for (int j = 0; j < 2; ++j) acc[i][j] = (floatx8)0.0f;

  float4 pav[4], pbv[4];
  auto load_regs = [&](int kk) {
    #pragma unroll
    for (int it = 0; it < 4; ++it) {
      int lin = tid + it * 256;          // 1024 float4 per tile
      int row = lin >> 3;
      int c4  = (lin & 7) * 4;
      float4 av = make_float4(0.f, 0.f, 0.f, 0.f);
      int gm = m0 + row;
      if (gm < M) av = *(const float4*)(A + (size_t)gm * K + kk + c4);
      pav[it] = av;
      float4 bv = make_float4(0.f, 0.f, 0.f, 0.f);
      int gn = n0 + row;
      if (gn < N) bv = *(const float4*)(W + (size_t)gn * K + kk + c4);
      pbv[it] = bv;
    }
  };
  auto store_lds = [&]() {
    #pragma unroll
    for (int it = 0; it < 4; ++it) {
      int lin = tid + it * 256;
      int row = lin >> 3;
      int c4  = (lin & 7) * 4;
      As[row][c4 + 0] = f2bf(pav[it].x); As[row][c4 + 1] = f2bf(pav[it].y);
      As[row][c4 + 2] = f2bf(pav[it].z); As[row][c4 + 3] = f2bf(pav[it].w);
      Bs[row][c4 + 0] = f2bf(pbv[it].x); Bs[row][c4 + 1] = f2bf(pbv[it].y);
      Bs[row][c4 + 2] = f2bf(pbv[it].z); Bs[row][c4 + 3] = f2bf(pbv[it].w);
    }
  };

  load_regs(0);
  store_lds();
  __syncthreads();

  for (int kk = 0; kk < K; kk += BK) {
    const bool more = (kk + BK) < K;
    if (more) load_regs(kk + BK);                 // loads in flight over compute
    if (kk + 2 * BK < K) {                        // prefetch 2 tiles ahead (L2)
      #pragma unroll
      for (int it = 0; it < 4; ++it) {
        int lin = tid + it * 256;
        int row = lin >> 3;
        int c4  = (lin & 7) * 4;
        int gm = m0 + row;
        if (gm < M) __builtin_prefetch(A + (size_t)gm * K + kk + 2 * BK + c4, 0, 1);
        int gn = n0 + row;
        if (gn < N) __builtin_prefetch(W + (size_t)gn * K + kk + 2 * BK + c4, 0, 1);
      }
    }
    wmma_tiles(As, Bs, acc, wr, wc, lane);
    __syncthreads();
    if (more) store_lds();
    __syncthreads();
  }

  // ---- epilogue (C/D layout: lane = N col, vgpr r = M row, +8 for hi half) ----
  const int nloc = lane & 15;
  const int mh   = (lane >> 4) * 8;
  #pragma unroll
  for (int ri = 0; ri < 4; ++ri) {
    #pragma unroll
    for (int cj = 0; cj < 2; ++cj) {
      int n = n0 + wc * 32 + cj * 16 + nloc;
      if (n >= N) continue;
      float bb = bias ? bias[n] : 0.0f;
      #pragma unroll
      for (int r = 0; r < 8; ++r) {
        int m = m0 + wr * 64 + ri * 16 + mh + r;
        if (m >= M) continue;
        float v = acc[ri][cj][r] + bb;
        if (act == 1) v = silu_f(v);
        if (gate) v = resid[(size_t)m * N + n] + gate[(m / Tb) * gateStride + n] * v;
        C[(size_t)m * N + n] = v;
      }
    }
  }
}

// =====================================================================
// Implicit-GEMM 5x5 SAME conv via WMMA: M = B*H*W pixels, N = Cout,
// K = Cin*25; im2col materialized on the fly into the LDS A-tile.
// =====================================================================
__global__ __launch_bounds__(256) void conv_gemm_kernel(
    const float* __restrict__ in, const float* __restrict__ w,
    const float* __restrict__ bias, float* __restrict__ out,
    int Cin, int Cout)
{
  __shared__ uint16_t As[BM][LDT];
  __shared__ uint16_t Bs[BN][LDT];
  const int K = Cin * 25;
  const int tid  = threadIdx.x;
  const int lane = tid & 31;
  const int wave = tid >> 5;
  const int wr = wave >> 2;
  const int wc = wave & 3;
  const int m0 = blockIdx.y * BM;        // pixel tile
  const int n0 = blockIdx.x * BN;        // cout tile

  floatx8 acc[4][2];
  #pragma unroll
  for (int i = 0; i < 4; ++i)
    #pragma unroll
    for (int j = 0; j < 2; ++j) acc[i][j] = (floatx8)0.0f;

  for (int kk = 0; kk < K; kk += BK) {
    #pragma unroll
    for (int it = 0; it < 16; ++it) {
      int lin = tid + it * 256;          // 4096 elements each for A and B
      int row = lin >> 5;
      int kc  = lin & 31;
      int k   = kk + kc;
      float a = 0.f;
      {
        int m = m0 + row;
        int b = m >> 12, p = m & 4095;
        int yy = p >> 6, xx = p & 63;
        if (k < K) {
          int ci = k / 25, rr = k % 25;
          int iy = yy + rr / 5 - 2, ix = xx + rr % 5 - 2;
          if (iy >= 0 && iy < IMG && ix >= 0 && ix < IMG)
            a = in[((size_t)(b * Cin + ci) * IMG + iy) * IMG + ix];
        }
      }
      As[row][kc] = f2bf(a);
      float bv = 0.f;
      int n = n0 + row;
      if (k < K && n < Cout) bv = w[(size_t)n * K + k];
      Bs[row][kc] = f2bf(bv);
    }
    __syncthreads();
    wmma_tiles(As, Bs, acc, wr, wc, lane);
    __syncthreads();
  }

  const int nloc = lane & 15;
  const int mh   = (lane >> 4) * 8;
  #pragma unroll
  for (int ri = 0; ri < 4; ++ri) {
    #pragma unroll
    for (int cj = 0; cj < 2; ++cj) {
      int co = n0 + wc * 32 + cj * 16 + nloc;
      if (co >= Cout) continue;
      float bb = bias[co];
      #pragma unroll
      for (int r = 0; r < 8; ++r) {
        int m = m0 + wr * 64 + ri * 16 + mh + r;
        int b = m >> 12, p = m & 4095;
        out[(size_t)(b * Cout + co) * (IMG * IMG) + p] = acc[ri][cj][r] + bb;
      }
    }
  }
}

// =====================================================================
// Attention: one wave per (b, h, 16-query tile). K/V tiles staged into
// LDS with global_load_async_to_lds_b128 (ASYNCcnt, ping-pong, one
// tile in flight); fragments built from LDS; S=QK^T and O=PV on WMMA.
// smem: S 64KB | P 32KB | KV staging 16KB  = 112KB
// =====================================================================
__global__ void attn_kernel(const float* __restrict__ q, const float* __restrict__ k,
                            const float* __restrict__ v, float* __restrict__ o,
                            float scale)
{
  extern __shared__ char smem[];
  float*    S  = (float*)smem;                                  // 16 x 1024 f32
  uint16_t* P  = (uint16_t*)(smem + 16 * TOKN * sizeof(float)); // 16 x 1024 bf16
  char*     KV = smem + 16 * TOKN * 6;                          // 16KB staging
  const uint32_t kv_off = (uint32_t)(uintptr_t)KV;              // LDS byte offset
  const int lane = threadIdx.x;
  const int m0 = blockIdx.x * 16;
  const int h  = blockIdx.y;
  const int b  = blockIdx.z;
  const size_t base = (size_t)(b * TOKN) * DIMM + h * HD;

  // Q fragments (A 16x32, two K-halves of hd=64)
  bf16x16 qa[2];
  {
    int m  = m0 + (lane & 15);
    int kb = (lane >> 4) * 16;
    const float* qrow = q + base + (size_t)m * DIMM;
    #pragma unroll
    for (int s = 0; s < 2; ++s) {
      union { uint32_t u[8]; bf16x16 vv; } f;
      #pragma unroll
      for (int j = 0; j < 4; ++j) {
        int k0 = s * 32 + kb + 2 * j;
        f.u[j]     = pack_bf16x2(qrow[k0], qrow[k0 + 1]);
        int k1 = s * 32 + 8 + kb + 2 * j;
        f.u[4 + j] = pack_bf16x2(qrow[k1], qrow[k1 + 1]);
      }
      qa[s] = f.vv;
    }
  }

  // ---- stage helpers: copy key/value rows (256B each) into LDS ----
  // K-tile: 16 rows x 64 f32 = 4KB  (8 b128 issues/lane)
  auto stage_k16 = [&](int n0, int buf) {
    const char* src = (const char*)(k + base + (size_t)n0 * DIMM);
    uint32_t dst = kv_off + buf * 4096;
    #pragma unroll
    for (int i = 0; i < 8; ++i) {
      uint32_t off = (uint32_t)(i * 32 + lane) * 16;
      uint32_t row = off >> 8, colb = off & 255;
      async_g2l_b128(dst + off, src + (size_t)row * (DIMM * 4) + colb);
    }
  };
  // V-tile: 32 rows x 64 f32 = 8KB  (16 b128 issues/lane)
  auto stage_v32 = [&](int ko, int buf) {
    const char* src = (const char*)(v + base + (size_t)ko * DIMM);
    uint32_t dst = kv_off + buf * 8192;
    #pragma unroll
    for (int i = 0; i < 16; ++i) {
      uint32_t off = (uint32_t)(i * 32 + lane) * 16;
      uint32_t row = off >> 8, colb = off & 255;
      async_g2l_b128(dst + off, src + (size_t)row * (DIMM * 4) + colb);
    }
  };

  // ---- S = scale * Q K^T  (64 key tiles of 16, pipelined async) ----
  stage_k16(0, 0);
  for (int ti = 0; ti < TOKN / 16; ++ti) {
    const bool more = (ti + 1) < TOKN / 16;
    if (more) stage_k16((ti + 1) * 16, (ti + 1) & 1);
    if (more) wait_async_le1(); else wait_async_0();
    const float* Kv = (const float*)(KV + (ti & 1) * 4096);
    floatx8 acc = (floatx8)0.0f;
    int key = lane & 15;
    #pragma unroll
    for (int s = 0; s < 2; ++s) {
      union { uint32_t u[8]; bf16x16 vv; } f;
      int kdb = s * 32 + (lane >> 4) * 16;
      #pragma unroll
      for (int j = 0; j < 8; ++j)
        f.u[j] = pack_bf16x2(Kv[key * 64 + kdb + 2 * j], Kv[key * 64 + kdb + 2 * j + 1]);
      acc = __builtin_amdgcn_wmma_f32_16x16x32_bf16(
          false, qa[s], false, f.vv, (short)0, acc, false, false);
    }
    int nn = ti * 16 + key;
    int mb = (lane >> 4) * 8;
    #pragma unroll
    for (int r = 0; r < 8; ++r) S[(mb + r) * TOKN + nn] = acc[r] * scale;
  }
  __syncthreads();

  // ---- softmax: 2 lanes per row (wave32 halves via shfl_xor 16) ----
  {
    int rrow = lane & 15, half = lane >> 4;
    const float* Sr = S + rrow * TOKN + half * 512;
    float mx = -3.4e38f;
    for (int j = 0; j < 512; ++j) mx = fmaxf(mx, Sr[j]);
    mx = fmaxf(mx, __shfl_xor(mx, 16, 32));
    float sum = 0.f;
    for (int j = 0; j < 512; ++j) sum += __expf(Sr[j] - mx);
    sum += __shfl_xor(sum, 16, 32);
    float inv = 1.0f / sum;
    uint16_t* Pr = P + rrow * TOKN + half * 512;
    for (int j = 0; j < 512; ++j) Pr[j] = f2bf(__expf(Sr[j] - mx) * inv);
  }
  __syncthreads();

  // ---- O = P V  (32 key tiles of 32, pipelined async) ----
  floatx8 oacc[4];
  #pragma unroll
  for (int c = 0; c < 4; ++c) oacc[c] = (floatx8)0.0f;
  stage_v32(0, 0);
  for (int ti = 0; ti < TOKN / 32; ++ti) {
    const bool more = (ti + 1) < TOKN / 32;
    if (more) stage_v32((ti + 1) * 32, (ti + 1) & 1);
    if (more) wait_async_le1(); else wait_async_0();
    const float* Vv = (const float*)(KV + (ti & 1) * 8192);
    int ko = ti * 32;
    union { uint32_t u[8]; bf16x16 vv; } fa;
    {
      int m  = lane & 15;
      int kb = (lane >> 4) * 16;
      const uint16_t* Pr = P + m * TOKN + ko;
      #pragma unroll
      for (int j = 0; j < 4; ++j) {
        fa.u[j]     = *(const uint32_t*)&Pr[kb + 2 * j];
        fa.u[4 + j] = *(const uint32_t*)&Pr[8 + kb + 2 * j];
      }
    }
    #pragma unroll
    for (int c = 0; c < 4; ++c) {
      union { uint32_t u[8]; bf16x16 vv; } fb;
      int col = c * 16 + (lane & 15);
      int kr  = (lane >> 4) * 16;          // row within 32-row tile
      #pragma unroll
      for (int j = 0; j < 8; ++j)
        fb.u[j] = pack_bf16x2(Vv[(kr + 2 * j) * 64 + col], Vv[(kr + 2 * j + 1) * 64 + col]);
      oacc[c] = __builtin_amdgcn_wmma_f32_16x16x32_bf16(
          false, fa.vv, false, fb.vv, (short)0, oacc[c], false, false);
    }
  }
  {
    int nn = lane & 15;
    int mb = (lane >> 4) * 8;
    #pragma unroll
    for (int c = 0; c < 4; ++c)
      #pragma unroll
      for (int r = 0; r < 8; ++r)
        o[base + (size_t)(m0 + mb + r) * DIMM + c * 16 + nn] = oacc[c][r];
  }
}

// ===================== norm / pointwise kernels ======================
__global__ void gn_stats_kernel(const float* __restrict__ x, float* mu, float* rstd)
{
  __shared__ float s1[256], s2[256];
  const int bg = blockIdx.x;                 // b*32 + g
  const int b = bg >> 5, g = bg & 31;
  const int tid = threadIdx.x;
  const int chPer = CONV_C / GN_G;           // 16
  const int HW = IMG * IMG;
  float sum = 0.f, sq = 0.f;
  for (int i = tid; i < chPer * HW; i += 256) {
    int c = i / HW, p = i % HW;
    float s = silu_f(x[(size_t)(b * CONV_C + g * chPer + c) * HW + p]);
    sum += s; sq += s * s;
  }
  s1[tid] = sum; s2[tid] = sq; __syncthreads();
  for (int st = 128; st > 0; st >>= 1) {
    if (tid < st) { s1[tid] += s1[tid + st]; s2[tid] += s2[tid + st]; }
    __syncthreads();
  }
  if (tid == 0) {
    float m = s1[0] / (chPer * HW);
    float var = s2[0] / (chPer * HW) - m * m;
    mu[bg] = m; rstd[bg] = rsqrtf(var + 1e-5f);
  }
}

__global__ void gn_apply_kernel(const float* x, const float* mu, const float* rstd,
                                const float* gamma, const float* beta, float* out)
{
  int idx = blockIdx.x * 256 + threadIdx.x;
  const int HW = IMG * IMG;
  int bc = idx / HW; int c = bc % CONV_C; int b = bc / CONV_C;
  int bg = b * GN_G + c / (CONV_C / GN_G);
  float s = silu_f(x[idx]);
  out[idx] = (s - mu[bg]) * rstd[bg] * gamma[c] + beta[c];
}

__global__ void patchify_kernel(const float* __restrict__ in, float* __restrict__ tok)
{
  int idx = blockIdx.x * 256 + threadIdx.x;          // B*1024*2048
  int f = idx % 2048; int bt = idx / 2048; int t = bt % TOKN; int b = bt / TOKN;
  int c = f >> 2; int py = (f >> 1) & 1; int px = f & 1;
  int ph = t >> 5, pw = t & 31;
  tok[idx] = in[((size_t)(b * CONV_C + c) * IMG + (ph * 2 + py)) * IMG + (pw * 2 + px)];
}

__global__ void temb1_kernel(const float* t, const float* w, const float* bias, float* out)
{
  int idx = blockIdx.x * 256 + threadIdx.x;          // B*DIM
  int d = idx % DIMM, b = idx / DIMM;
  out[idx] = silu_f(t[b] * w[d] + bias[d]);
}

__global__ void cond_kernel(const float* temb1, const float* w, const float* bias,
                            const float* yemb, const int* y, float* csilu)
{
  int idx = blockIdx.x * 256 + threadIdx.x;          // B*DIM
  int d = idx % DIMM, b = idx / DIMM;
  const float* xr = temb1 + (size_t)b * DIMM;
  const float* wr = w + (size_t)d * DIMM;
  float s = 0.f;
  for (int j = 0; j < DIMM; ++j) s += xr[j] * wr[j];
  s += bias[d] + yemb[(size_t)y[b] * DIMM + d];
  csilu[idx] = silu_f(s);
}

__global__ void gemv_kernel(const float* x, const float* W, const float* bias,
                            float* out, int K, int N)
{
  int n = blockIdx.x * 256 + threadIdx.x;
  int b = blockIdx.y;
  if (n >= N) return;
  const float* xr = x + (size_t)b * K;
  const float* wr = W + (size_t)n * K;
  float s = 0.f;
  for (int j = 0; j < K; ++j) s += xr[j] * wr[j];
  out[(size_t)b * N + n] = s + (bias ? bias[n] : 0.f);
}

__global__ void ln_mod_kernel(const float* __restrict__ x, const float* __restrict__ g,
                              const float* __restrict__ bta, const float* __restrict__ shift,
                              const float* __restrict__ scl, int modStride,
                              float* __restrict__ out, float eps)
{
  __shared__ float s1[256], s2[256];
  const int row = blockIdx.x;                // B*TOKN rows
  const int bi = row / TOKN;
  const int tid = threadIdx.x;
  const float* xr = x + (size_t)row * DIMM;
  float sum = 0.f, sq = 0.f;
  for (int j = tid; j < DIMM; j += 256) { float v = xr[j]; sum += v; sq += v * v; }
  s1[tid] = sum; s2[tid] = sq; __syncthreads();
  for (int st = 128; st > 0; st >>= 1) {
    if (tid < st) { s1[tid] += s1[tid + st]; s2[tid] += s2[tid + st]; }
    __syncthreads();
  }
  float m = s1[0] / DIMM;
  float rstd = rsqrtf(s2[0] / DIMM - m * m + eps);
  const float* sh = shift + (size_t)bi * modStride;
  const float* sc = scl + (size_t)bi * modStride;
  float* orow = out + (size_t)row * DIMM;
  for (int j = tid; j < DIMM; j += 256) {
    float xn = (xr[j] - m) * rstd;
    if (g) xn = xn * g[j] + bta[j];
    orow[j] = xn * (1.0f + sc[j]) + sh[j];
  }
}

__global__ void rope_kernel(float* q, float* k)
{
  int idx = blockIdx.x * 256 + threadIdx.x;  // B*T*H*32 pairs
  int i = idx & 31; int rem = idx >> 5;
  int h = rem % HEADS; int bt = rem / HEADS; int t = bt % TOKN;
  float fr = __expf(-(float)(2 * i) * 0.015625f * 9.210340371976184f); // 10000^(-2i/64)
  float f = (float)t * fr;
  float cs = __cosf(f), sn = __sinf(f);
  size_t off = (size_t)bt * DIMM + h * HD + 2 * i;
  float q0 = q[off], q1 = q[off + 1];
  q[off]     = q0 * cs - q1 * sn;
  q[off + 1] = q0 * sn + q1 * cs;
  float k0 = k[off], k1 = k[off + 1];
  k[off]     = k0 * cs - k1 * sn;
  k[off + 1] = k0 * sn + k1 * cs;
}

__global__ void mul_kernel(float* U, const float* V, int n)
{
  int i = blockIdx.x * 256 + threadIdx.x;
  if (i < n) U[i] *= V[i];
}

__global__ void unpatchify_kernel(const float* __restrict__ fin, float* __restrict__ out)
{
  int idx = blockIdx.x * 256 + threadIdx.x;  // B*CIN*64*64
  int gx = idx & 63; int rem = idx >> 6; int gy = rem & 63; rem >>= 6;
  int ci = rem & 3; int b = rem >> 2;
  int ph = gy >> 1, py = gy & 1, pw = gx >> 1, px = gx & 1;
  out[idx] = fin[(size_t)(b * TOKN + ph * 32 + pw) * 16 + (py * 2 + px) * CINCH + ci];
}

// =============================== launcher ===============================
static inline void gemm(const float* A, const float* W, float* C,
                        int M, int N, int K, const float* bias, int act,
                        const float* gate, int gs, const float* resid,
                        hipStream_t s)
{
  dim3 grid((N + BN - 1) / BN, (M + BM - 1) / BM);
  gemm_bf16_kernel<<<grid, 256, 0, s>>>(A, W, C, M, N, K, bias, act, gate, gs, TOKN, resid);
}

extern "C" void kernel_launch(void* const* d_in, const int* in_sizes, int n_in,
                              void* d_out, int out_size, void* d_ws, size_t ws_size,
                              hipStream_t stream)
{
  (void)in_sizes; (void)n_in; (void)out_size; (void)ws_size;
  const float* x        = (const float*)d_in[0];
  const float* t        = (const float*)d_in[1];
  const int*   y        = (const int*)  d_in[2];
  const float* conv1_w  = (const float*)d_in[3];
  const float* conv1_b  = (const float*)d_in[4];
  const float* gn1_g    = (const float*)d_in[5];
  const float* gn1_b    = (const float*)d_in[6];
  const float* conv2_w  = (const float*)d_in[7];
  const float* conv2_b  = (const float*)d_in[8];
  const float* gn2_g    = (const float*)d_in[9];
  const float* gn2_b    = (const float*)d_in[10];
  const float* xemb_w   = (const float*)d_in[11];
  const float* xemb_b   = (const float*)d_in[12];
  const float* t1_w     = (const float*)d_in[13];
  const float* t1_b     = (const float*)d_in[14];
  const float* t2_w     = (const float*)d_in[15];
  const float* t2_b     = (const float*)d_in[16];
  const float* y_emb    = (const float*)d_in[17];
  const float* wq       = (const float*)d_in[18];
  const float* wk       = (const float*)d_in[19];
  const float* wv       = (const float*)d_in[20];
  const float* wo       = (const float*)d_in[21];
  const float* w1       = (const float*)d_in[22];
  const float* w3       = (const float*)d_in[23];
  const float* w2       = (const float*)d_in[24];
  const float* ada_w    = (const float*)d_in[25];
  const float* ada_b    = (const float*)d_in[26];
  const float* an_g     = (const float*)d_in[27];
  const float* an_b     = (const float*)d_in[28];
  const float* fn_g     = (const float*)d_in[29];
  const float* fn_b     = (const float*)d_in[30];
  const float* fada_w   = (const float*)d_in[31];
  const float* fada_b   = (const float*)d_in[32];
  const float* fin_w    = (const float*)d_in[33];
  const float* fin_b    = (const float*)d_in[34];
  float* out = (float*)d_out;

  // ---- workspace arena (phase-aliased region R + persistent z/hh + smalls) ----
  char* ws = (char*)d_ws;
  const size_t MB = 1024u * 1024u;
  const size_t ACT  = (size_t)BATCH * TOKN * DIMM * sizeof(float);          // 16 MB
  const size_t CONVB = (size_t)BATCH * CONV_C * IMG * IMG * sizeof(float);  // 32 MB
  const size_t FFB  = (size_t)BATCH * TOKN * FFD * sizeof(float);           // 44 MB
  const size_t RB   = 100 * MB;                                             // aliased region
  float* bufA  = (float*)(ws + 0);            // conv ping   (also tok)
  float* bufB  = (float*)(ws + CONVB);        // conv pong
  float* tok   = bufA;                        // after patchify (reads bufB)
  float* qb    = (float*)(ws + 0);            // transformer phase aliases
  float* kb    = (float*)(ws + ACT);
  float* vb    = (float*)(ws + 2 * ACT);
  float* attno = (float*)(ws + 3 * ACT);
  float* U     = (float*)(ws + 0);            // FF phase aliases q..attno
  float* V     = (float*)(ws + FFB);
  float* z     = (float*)(ws + RB);
  float* hh    = (float*)(ws + RB + ACT);
  char*  sm    = ws + RB + 2 * ACT;
  float* temb1b = (float*)(sm + 0);
  float* csilu  = (float*)(sm + 16384);
  float* mods   = (float*)(sm + 32768);       // B x 6144
  float* fm     = (float*)(sm + 163840);      // B x 2048
  float* gmu    = (float*)(sm + 212992);
  float* grstd  = (float*)(sm + 213504);
  float* finbuf = (float*)(sm + 214016);      // 4096 x 16

  const int MPIX = BATCH * IMG * IMG;         // 16384 pixels

  // ---------------- stem (implicit-GEMM convs on WMMA) ----------------
  conv_gemm_kernel<<<dim3(CONV_C / BN, MPIX / BM), 256, 0, stream>>>(
      x, conv1_w, conv1_b, bufA, CINCH, CONV_C);
  gn_stats_kernel<<<BATCH * GN_G, 256, 0, stream>>>(bufA, gmu, grstd);
  gn_apply_kernel<<<(BATCH * CONV_C * IMG * IMG) / 256, 256, 0, stream>>>(
      bufA, gmu, grstd, gn1_g, gn1_b, bufB);
  conv_gemm_kernel<<<dim3(CONV_C / BN, MPIX / BM), 256, 0, stream>>>(
      bufB, conv2_w, conv2_b, bufA, CONV_C, CONV_C);
  gn_stats_kernel<<<BATCH * GN_G, 256, 0, stream>>>(bufA, gmu, grstd);
  gn_apply_kernel<<<(BATCH * CONV_C * IMG * IMG) / 256, 256, 0, stream>>>(
      bufA, gmu, grstd, gn2_g, gn2_b, bufB);
  patchify_kernel<<<(BATCH * TOKN * 2048) / 256, 256, 0, stream>>>(bufB, tok);

  // token embedding: z = tok @ xemb_w^T + xemb_b
  gemm(tok, xemb_w, z, BATCH * TOKN, DIMM, 2048, xemb_b, 0, nullptr, 0, nullptr, stream);

  // conditioning
  temb1_kernel<<<(BATCH * DIMM) / 256, 256, 0, stream>>>(t, t1_w, t1_b, temb1b);
  cond_kernel<<<(BATCH * DIMM) / 256, 256, 0, stream>>>(temb1b, t2_w, t2_b, y_emb, y, csilu);

  const int M = BATCH * TOKN;
  for (int i = 0; i < LAYERS; ++i) {
    const float* wq_i = wq + (size_t)i * DIMM * DIMM;
    const float* wk_i = wk + (size_t)i * DIMM * DIMM;
    const float* wv_i = wv + (size_t)i * DIMM * DIMM;
    const float* wo_i = wo + (size_t)i * DIMM * DIMM;
    const float* w1_i = w1 + (size_t)i * FFD * DIMM;
    const float* w3_i = w3 + (size_t)i * FFD * DIMM;
    const float* w2_i = w2 + (size_t)i * DIMM * FFD;

    gemv_kernel<<<dim3((6 * DIMM + 255) / 256, BATCH), 256, 0, stream>>>(
        csilu, ada_w + (size_t)i * 6 * DIMM * DIMM, ada_b + (size_t)i * 6 * DIMM,
        mods, DIMM, 6 * DIMM);

    ln_mod_kernel<<<M, 256, 0, stream>>>(
        z, an_g + i * DIMM, an_b + i * DIMM, mods + 0, mods + DIMM, 6 * DIMM, hh, 1e-5f);

    gemm(hh, wq_i, qb, M, DIMM, DIMM, nullptr, 0, nullptr, 0, nullptr, stream);
    gemm(hh, wk_i, kb, M, DIMM, DIMM, nullptr, 0, nullptr, 0, nullptr, stream);
    gemm(hh, wv_i, vb, M, DIMM, DIMM, nullptr, 0, nullptr, 0, nullptr, stream);
    rope_kernel<<<(BATCH * TOKN * HEADS * 32) / 256, 256, 0, stream>>>(qb, kb);
    attn_kernel<<<dim3(TOKN / 16, HEADS, BATCH), 32,
                  16 * TOKN * 6 + 16384, stream>>>(qb, kb, vb, attno, 0.125f);
    gemm(attno, wo_i, z, M, DIMM, DIMM, nullptr, 0, mods + 2 * DIMM, 6 * DIMM, z, stream);

    ln_mod_kernel<<<M, 256, 0, stream>>>(
        z, fn_g + i * DIMM, fn_b + i * DIMM, mods + 3 * DIMM, mods + 4 * DIMM, 6 * DIMM,
        hh, 1e-5f);
    gemm(hh, w1_i, U, M, FFD, DIMM, nullptr, 1, nullptr, 0, nullptr, stream);  // silu
    gemm(hh, w3_i, V, M, FFD, DIMM, nullptr, 0, nullptr, 0, nullptr, stream);
    mul_kernel<<<(M * FFD) / 256, 256, 0, stream>>>(U, V, M * FFD);
    gemm(U, w2_i, z, M, DIMM, FFD, nullptr, 0, mods + 5 * DIMM, 6 * DIMM, z, stream);
  }

  gemv_kernel<<<dim3((2 * DIMM + 255) / 256, BATCH), 256, 0, stream>>>(
      csilu, fada_w, fada_b, fm, DIMM, 2 * DIMM);
  ln_mod_kernel<<<M, 256, 0, stream>>>(
      z, nullptr, nullptr, fm + 0, fm + DIMM, 2 * DIMM, hh, 1e-6f);
  gemm(hh, fin_w, finbuf, M, 16, DIMM, fin_b, 0, nullptr, 0, nullptr, stream);
  unpatchify_kernel<<<(BATCH * CINCH * IMG * IMG) / 256, 256, 0, stream>>>(finbuf, out);
}
